// SwinTransformerBlock_20822001451660
// MI455X (gfx1250) — matrix operand order repaired and checked by reading, not compiled, not run on hardware
//
#include <hip/hip_runtime.h>
#include <math.h>

// ---------------------------------------------------------------------------
// CDNA5 (gfx1250) Swin block: GEMMs via v_wmma_f32_16x16x32_f16 (wave32) with
// double-buffered Tensor-Data-Mover (tensor_load_to_lds) LDS staging.
// ---------------------------------------------------------------------------

typedef __attribute__((ext_vector_type(16))) _Float16 v16h;
typedef __attribute__((ext_vector_type(8)))  _Float16 v8h;
typedef __attribute__((ext_vector_type(8)))  float    v8f;
typedef __attribute__((ext_vector_type(4)))  unsigned int v4u;
typedef __attribute__((ext_vector_type(8)))  int      v8i;
typedef __attribute__((ext_vector_type(4)))  int      v4i;

constexpr int kC     = 384;
constexpr int kHeads = 12;
constexpr int kHd    = 32;
constexpr int kB     = 16;
constexpr int kH     = 56;
constexpr int kW     = 56;
constexpr int kWS    = 7;
constexpr int kShift = 3;
constexpr int kNWin  = (kH / kWS) * (kW / kWS);   // 64 windows / image
constexpr int kTok   = kWS * kWS;                 // 49 tokens / window
constexpr int kWins  = kB * kNWin;                // 1024 windows
constexpr int kM     = kWins * kTok;              // 50176 rows (win,token order)
constexpr int kPix   = kB * kH * kW;              // 50176 pixels (b,i,j order)
constexpr int kHid   = 4 * kC;                    // 1536
constexpr int kQKV   = 3 * kC;                    // 1152
constexpr float kQScale = 0.17677669529663689f;   // 32^-0.5
constexpr float kEps    = 1e-5f;

__device__ inline v8f vzero() {
  v8f z;
  for (int i = 0; i < 8; ++i) z[i] = 0.0f;
  return z;
}

__device__ inline v8f wmma_f16(v16h a, v16h b, v8f c) {
  // D = A(16x32 f16) x B(32x16 f16) + C(16x16 f32)
  return __builtin_amdgcn_wmma_f32_16x16x32_f16(false, a, false, b,
                                                (short)0, c, false, false);
}

// A-fragment (16x32 f16): lane l -> row m0+(l&15); VGPRs 0..3 hold K = h*8..h*8+7,
// VGPRs 4..7 hold K = 16+h*8..+7 with h = l>>4.  Two 16-byte LDS reads.
__device__ inline v16h load_frag_a(const _Float16* base, int stride, int m0, int k0) {
  const int l = threadIdx.x & 31;
  const _Float16* p = base + (m0 + (l & 15)) * stride + k0 + (l >> 4) * 8;
  v8h lo = *(const v8h*)p;
  v8h hi = *(const v8h*)(p + 16);
  return __builtin_shufflevector(lo, hi, 0, 1, 2, 3, 4, 5, 6, 7,
                                 8, 9, 10, 11, 12, 13, 14, 15);
}

// B-fragment (32x16 f16) from an N-major [N][K] tile (i.e. W row-major):
// lane l -> col n0+(l&15), K = h*8..h*8+7 then 16+h*8..+7 with h = l>>4.
__device__ inline v16h load_frag_b(const _Float16* base, int stride, int n0, int k0) {
  const int l = threadIdx.x & 31;
  const _Float16* p = base + (n0 + (l & 15)) * stride + k0 + (l >> 4) * 8;
  v8h lo = *(const v8h*)p;
  v8h hi = *(const v8h*)(p + 16);
  return __builtin_shufflevector(lo, hi, 0, 1, 2, 3, 4, 5, 6, 7,
                                 8, 9, 10, 11, 12, 13, 14, 15);
}

// ---------------------------------------------------------------------------
// TDM: DMA a [128 rows x 32 f16] tile (row stride K elements) from global into
// LDS at lds_off, padding each 64B row with 16B so LDS row stride = 40 halves.
// D# layout per CDNA5 ISA ch.8 (group0: count/lds/global/type; group1: flags,
// dims, tile, stride).  2D tile -> groups 2/3 zero.
// ---------------------------------------------------------------------------
__device__ inline void tdm_load_tile(const _Float16* gptr, unsigned lds_off, int K) {
  const unsigned long long ga = (unsigned long long)(size_t)gptr;
  v4u g0;
  g0[0] = 1u;                                          // count=1, user mode
  g0[1] = lds_off;                                     // lds_addr (bytes)
  g0[2] = (unsigned)(ga & 0xffffffffu);                // global_addr[31:0]
  g0[3] = (unsigned)((ga >> 32) & 0x1ffffffu)          // global_addr[56:32]
        | (2u << 30);                                  // type = 2 ("image")
  v8i g1;
  g1[0] = (int)((1u << 16)                             // data_size = 2 bytes
              | (1u << 20)                             // pad_enable
              | (3u << 22)                             // pad_interval = 16 DW (64B)
              | (3u << 25));                           // pad_amount   = 4 DW (16B)
  g1[1] = (int)(((unsigned)K & 0xffffu) << 16);        // tensor_dim0 lo16
  g1[2] = (int)((((unsigned)K >> 16) & 0xffffu)        // tensor_dim0 hi16
              | (128u << 16));                         // tensor_dim1 lo16 = 128
  g1[3] = (int)(32u << 16);                            // tile_dim0 = 32
  g1[4] = (int)128u;                                   // tile_dim1 = 128, tile_dim2=0
  g1[5] = (int)(unsigned)K;                            // tensor_dim0_stride lo32
  g1[6] = 0;                                           // stride hi16, dim1_stride lo16
  g1[7] = 0;
  v4i z4 = {0, 0, 0, 0};
#if __clang_major__ >= 23
  v8i z8 = {0, 0, 0, 0, 0, 0, 0, 0};
  __builtin_amdgcn_tensor_load_to_lds(g0, g1, z4, z4, z8, 0);
#else
  __builtin_amdgcn_tensor_load_to_lds(g0, g1, z4, z4, 0);
#endif
}

// ---------------------------------------------------------------------------
// Generic tiled WMMA GEMM:  C[M,N] = A[M,K] * W[N,K]^T  (A,W f16, acc f32)
// BM=BN=128, BK=32, 256 threads = 8 waves, wave owns 32(M) x 64(N).
// Tiles staged by the Tensor Data Mover, double-buffered on TENSORcnt.
// ---------------------------------------------------------------------------
template <class Epi>
__global__ __launch_bounds__(256) void gemm_kernel(const _Float16* __restrict__ A,
                                                   const _Float16* __restrict__ Wt,
                                                   int K, Epi epi) {
  __shared__ _Float16 sA[2][128 * 40];  // row stride 40 halves = 80B (16B aligned)
  __shared__ _Float16 sB[2][128 * 40];
  const int m0  = blockIdx.x * 128;
  const int n0  = blockIdx.y * 128;
  const int tid = threadIdx.x;
  const int wv  = tid >> 5;
  const int wm  = (wv >> 1) * 32;       // wave M offset (4 wave-rows)
  const int wn  = (wv & 1) * 64;        // wave N offset (2 wave-cols)

  v8f acc[2][4];
#pragma unroll
  for (int i = 0; i < 2; ++i)
#pragma unroll
    for (int j = 0; j < 4; ++j) acc[i][j] = vzero();

  const _Float16* gA = A  + (size_t)m0 * K;
  const _Float16* gB = Wt + (size_t)n0 * K;
  const int nk = K >> 5;

  if (wv == 0) {  // wave 0 drives the TDM (EXEC-independent, wave-level DMA)
    tdm_load_tile(gA, (unsigned)(size_t)&sA[0][0], K);
    tdm_load_tile(gB, (unsigned)(size_t)&sB[0][0], K);
  }

  for (int ki = 0; ki < nk; ++ki) {
    const int cur = ki & 1;
    if (wv == 0) {
      if (ki + 1 < nk) {
        tdm_load_tile(gA + (ki + 1) * 32, (unsigned)(size_t)&sA[cur ^ 1][0], K);
        tdm_load_tile(gB + (ki + 1) * 32, (unsigned)(size_t)&sB[cur ^ 1][0], K);
        __builtin_amdgcn_s_wait_tensorcnt(2);  // current tiles resident
      } else {
        __builtin_amdgcn_s_wait_tensorcnt(0);
      }
    }
    __syncthreads();

    v16h af[2], bf[4];
#pragma unroll
    for (int i = 0; i < 2; ++i) af[i] = load_frag_a(sA[cur], 40, wm + i * 16, 0);
#pragma unroll
    for (int j = 0; j < 4; ++j) bf[j] = load_frag_b(sB[cur], 40, wn + j * 16, 0);
#pragma unroll
    for (int i = 0; i < 2; ++i)
#pragma unroll
      for (int j = 0; j < 4; ++j) acc[i][j] = wmma_f16(af[i], bf[j], acc[i][j]);
    __syncthreads();  // tile consumed before TDM may overwrite this buffer
  }

  // C/D layout: lane l -> n = l&15 ; VGPR r -> m = r + (l>>4)*8
  const int l = tid & 31, ln = l & 15, hf = l >> 4;
#pragma unroll
  for (int i = 0; i < 2; ++i)
#pragma unroll
    for (int j = 0; j < 4; ++j)
#pragma unroll
      for (int r = 0; r < 8; ++r)
        epi(m0 + wm + i * 16 + r + hf * 8, n0 + wn + j * 16 + ln, acc[i][j][r]);
}

// ------------------------------- epilogues ---------------------------------
struct EpiQKV {   // +bias, scale q, scatter to per-window-head q/k/v buffers
  const float* bias;
  _Float16 *qb, *kb, *vb;
  __device__ void operator()(int m, int n, float v) const {
    v += bias[n];
    const int part = n / kC;
    const int rem  = n - part * kC;
    const int head = rem >> 5, dd = rem & 31;
    const int win  = m / kTok, tok = m - win * kTok;
    const size_t idx =
        ((size_t)(win * kHeads + head) * kTok + tok) * kHd + dd;
    if (part == 0)      qb[idx] = (_Float16)(v * kQScale);
    else if (part == 1) kb[idx] = (_Float16)v;
    else                vb[idx] = (_Float16)v;
  }
};

struct EpiBias {  // +bias -> f32 (proj output, win/token order)
  const float* bias;
  float* out;
  __device__ void operator()(int m, int n, float v) const {
    out[(size_t)m * kC + n] = v + bias[n];
  }
};

struct EpiGelu {  // +bias, exact GELU -> f16 hidden
  const float* bias;
  _Float16* out;
  __device__ void operator()(int m, int n, float v) const {
    v += bias[n];
    const float g = 0.5f * v * (1.0f + erff(v * 0.70710678118654752f));
    out[(size_t)m * kHid + n] = (_Float16)g;
  }
};

struct EpiOut {   // +bias, +residual x1, NHWC -> NCHW into d_out
  const float* bias;
  const float* x1;
  float* out;
  __device__ void operator()(int m, int n, float v) const {
    v += bias[n] + x1[(size_t)m * kC + n];
    const int b = m / (kH * kW);
    const int rem = m - b * (kH * kW);
    const int i = rem / kW, j = rem - i * kW;
    out[(((size_t)b * kC + n) * kH + i) * kW + j] = v;
  }
};

// ---------------------------------------------------------------------------
// Weight f32 -> f16 conversion
// ---------------------------------------------------------------------------
__global__ void cvt_kernel(const float* __restrict__ src,
                           _Float16* __restrict__ dst, int n) {
  int i = blockIdx.x * 256 + threadIdx.x;
  if (i < n) dst[i] = (_Float16)src[i];
}

// ---------------------------------------------------------------------------
// LN1 + cyclic shift + window partition. One wave per pixel.
// Writes f32 shortcut (pixel order, NHWC) and f16 GEMM operand (win/token).
// ---------------------------------------------------------------------------
__global__ __launch_bounds__(256) void ln1_kernel(const float* __restrict__ x,
                                                  const float* __restrict__ w,
                                                  const float* __restrict__ bsc,
                                                  float* __restrict__ xT,
                                                  _Float16* __restrict__ xw) {
  const int pix = blockIdx.x * 8 + (threadIdx.x >> 5);
  const int l = threadIdx.x & 31;
  const int b  = pix / (kH * kW);
  const int rm = pix - b * (kH * kW);
  const int hh = rm / kW, ww = rm - hh * kW;
  const float* xp = x + (size_t)b * kC * kH * kW + (size_t)hh * kW + ww;

  float vals[12];
  float s = 0.0f;
#pragma unroll
  for (int ii = 0; ii < 12; ++ii) {
    float t = xp[(size_t)(l + ii * 32) * (kH * kW)];
    vals[ii] = t;
    s += t;
  }
  for (int o = 16; o; o >>= 1) s += __shfl_xor(s, o, 32);
  const float mu = s * (1.0f / kC);
  float vs = 0.0f;
#pragma unroll
  for (int ii = 0; ii < 12; ++ii) {
    float d = vals[ii] - mu;
    vs += d * d;
  }
  for (int o = 16; o; o >>= 1) vs += __shfl_xor(vs, o, 32);
  const float inv = rsqrtf(vs * (1.0f / kC) + kEps);

  int si = hh - kShift; if (si < 0) si += kH;
  int sj = ww - kShift; if (sj < 0) sj += kW;
  const int win = (si / kWS) * (kW / kWS) + (sj / kWS);
  const int tok = (si % kWS) * kWS + (sj % kWS);
  float*    xtp = xT + (size_t)pix * kC;
  _Float16* xwp = xw + ((size_t)(b * kNWin + win) * kTok + tok) * kC;
#pragma unroll
  for (int ii = 0; ii < 12; ++ii) {
    const int c = l + ii * 32;
    const float t = vals[ii];
    xtp[c] = t;
    xwp[c] = (_Float16)((t - mu) * inv * w[c] + bsc[c]);
  }
}

// ---------------------------------------------------------------------------
// Windowed attention: one block per (window, head), 4 waves.
// S = q k^T via 4x4 WMMA tiles (49 padded to 64), masked softmax in C-frags,
// O = P v via WMMA, stored straight into proj-GEMM input layout.
// ---------------------------------------------------------------------------
__global__ __launch_bounds__(128) void attn_kernel(const _Float16* __restrict__ qb,
                                                   const _Float16* __restrict__ kb,
                                                   const _Float16* __restrict__ vb,
                                                   _Float16* __restrict__ pin) {
  const int wh   = blockIdx.x;            // 0 .. kWins*kHeads-1
  const int win  = wh / kHeads;
  const int head = wh - win * kHeads;
  __shared__ _Float16 qs[64 * 40];        // tokens x d   (padded)
  __shared__ _Float16 ks[64 * 40];        // tokens x d   (padded)
  __shared__ _Float16 vT[32 * 72];        // d x tokens   (padded)
  __shared__ _Float16 Ps[64 * 72];        // tokens x tokens (padded)
  const int tid = threadIdx.x;

  for (int i = tid; i < 64 * 40; i += 128) { qs[i] = (_Float16)0.0f; ks[i] = (_Float16)0.0f; }
  for (int i = tid; i < 32 * 72; i += 128) vT[i] = (_Float16)0.0f;
  __syncthreads();

  const _Float16* qg = qb + (size_t)wh * kTok * kHd;
  const _Float16* kg = kb + (size_t)wh * kTok * kHd;
  const _Float16* vg = vb + (size_t)wh * kTok * kHd;
  for (int cc = tid; cc < kTok * 4; cc += 128) {   // 16B chunks, rows of 32 f16
    const int t = cc >> 2, d8 = (cc & 3) * 8;
    *(uint4*)(&qs[t * 40 + d8]) = *(const uint4*)(&qg[t * kHd + d8]);
    *(uint4*)(&ks[t * 40 + d8]) = *(const uint4*)(&kg[t * kHd + d8]);
  }
  for (int e = tid; e < kTok * kHd; e += 128) {    // transpose v -> vT
    const int t = e >> 5, d = e & 31;
    vT[d * 72 + t] = vg[e];
  }
  __syncthreads();

  const int wv = tid >> 5, l = tid & 31;
  const int m0 = wv * 16;
  const int ln = l & 15, hf = l >> 4;

  // ---- S = q k^T : wave wv owns S rows m0..m0+15, 4 col tiles
  v16h aq = load_frag_a(qs, 40, m0, 0);
  v8f sfr[4];
#pragma unroll
  for (int t = 0; t < 4; ++t) {
    v16h bk = load_frag_b(ks, 40, t * 16, 0);
    sfr[t] = wmma_f16(aq, bk, vzero());
  }

  // ---- masked softmax over 49 cols, rows reduced across the 16-lane halves
#pragma unroll
  for (int r = 0; r < 8; ++r) {
    float mx = -1e30f;
#pragma unroll
    for (int t = 0; t < 4; ++t) {
      const int n = t * 16 + ln;
      if (n < kTok) mx = fmaxf(mx, sfr[t][r]);
    }
    for (int o = 8; o; o >>= 1) mx = fmaxf(mx, __shfl_xor(mx, o, 32));
    float sum = 0.0f;
#pragma unroll
    for (int t = 0; t < 4; ++t) {
      const int n = t * 16 + ln;
      const float e = (n < kTok) ? __expf(sfr[t][r] - mx) : 0.0f;
      sfr[t][r] = e;
      sum += e;
    }
    for (int o = 8; o; o >>= 1) sum += __shfl_xor(sum, o, 32);
    const float is = 1.0f / sum;
    const int m = m0 + r + hf * 8;
#pragma unroll
    for (int t = 0; t < 4; ++t) Ps[m * 72 + t * 16 + ln] = (_Float16)(sfr[t][r] * is);
  }
  __syncthreads();

  // ---- O = P v : 2 k-steps (K=64), 2 n-tiles (d=32)
  v8f oacc[2] = {vzero(), vzero()};
#pragma unroll
  for (int kk = 0; kk < 64; kk += 32) {
    v16h ap = load_frag_a(Ps, 72, m0, kk);
#pragma unroll
    for (int jt = 0; jt < 2; ++jt) {
      v16h bv = load_frag_b(vT, 72, jt * 16, kk);
      oacc[jt] = wmma_f16(ap, bv, oacc[jt]);
    }
  }
#pragma unroll
  for (int jt = 0; jt < 2; ++jt)
#pragma unroll
    for (int r = 0; r < 8; ++r) {
      const int m = m0 + r + hf * 8;
      if (m < kTok) {
        const int col = head * kHd + jt * 16 + ln;
        pin[((size_t)win * kTok + m) * kC + col] = (_Float16)oacc[jt][r];
      }
    }
}

// ---------------------------------------------------------------------------
// residual1 (un-shift + shortcut) fused with LN2. One wave per pixel.
// ---------------------------------------------------------------------------
__global__ __launch_bounds__(256) void ln2_kernel(const float* __restrict__ projo,
                                                  const float* __restrict__ xT,
                                                  const float* __restrict__ w,
                                                  const float* __restrict__ bsc,
                                                  float* __restrict__ x1,
                                                  _Float16* __restrict__ hin) {
  const int pix = blockIdx.x * 8 + (threadIdx.x >> 5);
  const int l = threadIdx.x & 31;
  const int b  = pix / (kH * kW);
  const int rm = pix - b * (kH * kW);
  const int i  = rm / kW, j = rm - i * kW;
  int si = i - kShift; if (si < 0) si += kH;
  int sj = j - kShift; if (sj < 0) sj += kW;
  const int win = (si / kWS) * (kW / kWS) + (sj / kWS);
  const int tok = (si % kWS) * kWS + (sj % kWS);
  const float* pp = projo + ((size_t)(b * kNWin + win) * kTok + tok) * kC;
  const float* xp = xT + (size_t)pix * kC;

  float vals[12];
  float s = 0.0f;
#pragma unroll
  for (int ii = 0; ii < 12; ++ii) {
    const int c = l + ii * 32;
    float t = xp[c] + pp[c];
    vals[ii] = t;
    s += t;
  }
  for (int o = 16; o; o >>= 1) s += __shfl_xor(s, o, 32);
  const float mu = s * (1.0f / kC);
  float vs = 0.0f;
#pragma unroll
  for (int ii = 0; ii < 12; ++ii) {
    float d = vals[ii] - mu;
    vs += d * d;
  }
  for (int o = 16; o; o >>= 1) vs += __shfl_xor(vs, o, 32);
  const float inv = rsqrtf(vs * (1.0f / kC) + kEps);

  float*    x1p = x1 + (size_t)pix * kC;
  _Float16* hp  = hin + (size_t)pix * kC;
#pragma unroll
  for (int ii = 0; ii < 12; ++ii) {
    const int c = l + ii * 32;
    const float t = vals[ii];
    x1p[c] = t;
    hp[c]  = (_Float16)((t - mu) * inv * w[c] + bsc[c]);
  }
}

// ---------------------------------------------------------------------------
extern "C" void kernel_launch(void* const* d_in, const int* in_sizes, int n_in,
                              void* d_out, int out_size, void* d_ws, size_t ws_size,
                              hipStream_t stream) {
  (void)in_sizes; (void)n_in; (void)out_size; (void)ws_size;
  const float* x       = (const float*)d_in[0];
  const float* n1w     = (const float*)d_in[1];
  const float* n1b     = (const float*)d_in[2];
  const float* qkv_w   = (const float*)d_in[3];
  const float* qkv_b   = (const float*)d_in[4];
  const float* proj_w  = (const float*)d_in[5];
  const float* proj_b  = (const float*)d_in[6];
  const float* n2w     = (const float*)d_in[7];
  const float* n2b     = (const float*)d_in[8];
  const float* mlp_w1  = (const float*)d_in[9];
  const float* mlp_b1  = (const float*)d_in[10];
  const float* mlp_w2  = (const float*)d_in[11];
  const float* mlp_b2  = (const float*)d_in[12];

  char* wsb = (char*)d_ws;
  size_t off = 0;
  auto nxt = [&](size_t bytes) -> void* {
    void* p = wsb + off;
    off += (bytes + 255) & ~(size_t)255;
    return p;
  };
  _Float16* wq   = (_Float16*)nxt((size_t)kQKV * kC * 2);
  _Float16* wp   = (_Float16*)nxt((size_t)kC * kC * 2);
  _Float16* w1   = (_Float16*)nxt((size_t)kHid * kC * 2);
  _Float16* w2m  = (_Float16*)nxt((size_t)kC * kHid * 2);
  _Float16* xw   = (_Float16*)nxt((size_t)kM * kC * 2);
  float*    xT   = (float*)   nxt((size_t)kPix * kC * 4);
  _Float16* qbuf = (_Float16*)nxt((size_t)kWins * kHeads * kTok * kHd * 2);
  _Float16* kbuf = (_Float16*)nxt((size_t)kWins * kHeads * kTok * kHd * 2);
  _Float16* vbuf = (_Float16*)nxt((size_t)kWins * kHeads * kTok * kHd * 2);
  _Float16* pin  = (_Float16*)nxt((size_t)kM * kC * 2);
  float*    projo= (float*)   nxt((size_t)kM * kC * 4);
  float*    x1   = (float*)   nxt((size_t)kPix * kC * 4);
  _Float16* hin  = (_Float16*)nxt((size_t)kPix * kC * 2);
  _Float16* hmid = (_Float16*)nxt((size_t)kPix * kHid * 2);

  // 1. weights -> f16
  cvt_kernel<<<(kQKV * kC + 255) / 256, 256, 0, stream>>>(qkv_w, wq, kQKV * kC);
  cvt_kernel<<<(kC * kC + 255) / 256, 256, 0, stream>>>(proj_w, wp, kC * kC);
  cvt_kernel<<<(kHid * kC + 255) / 256, 256, 0, stream>>>(mlp_w1, w1, kHid * kC);
  cvt_kernel<<<(kC * kHid + 255) / 256, 256, 0, stream>>>(mlp_w2, w2m, kC * kHid);

  // 2. LN1 + shift + window partition
  ln1_kernel<<<kPix / 8, 256, 0, stream>>>(x, n1w, n1b, xT, xw);

  // 3. QKV GEMM  (50176 x 384) x (384 x 1152)
  gemm_kernel<<<dim3(kM / 128, kQKV / 128), 256, 0, stream>>>(
      xw, wq, kC, EpiQKV{qkv_b, qbuf, kbuf, vbuf});

  // 4. windowed attention
  attn_kernel<<<kWins * kHeads, 128, 0, stream>>>(qbuf, kbuf, vbuf, pin);

  // 5. proj GEMM  (50176 x 384) x (384 x 384)
  gemm_kernel<<<dim3(kM / 128, kC / 128), 256, 0, stream>>>(
      pin, wp, kC, EpiBias{proj_b, projo});

  // 6. un-shift + residual + LN2
  ln2_kernel<<<kPix / 8, 256, 0, stream>>>(projo, xT, n2w, n2b, x1, hin);

  // 7. MLP1 GEMM + GELU  (50176 x 384) x (384 x 1536)
  gemm_kernel<<<dim3(kPix / 128, kHid / 128), 256, 0, stream>>>(
      hin, w1, kC, EpiGelu{mlp_b1, hmid});

  // 8. MLP2 GEMM + residual + NCHW store  (50176 x 1536) x (1536 x 384)
  gemm_kernel<<<dim3(kPix / 128, kC / 128), 256, 0, stream>>>(
      hmid, w2m, kHid, EpiOut{mlp_b2, x1, (float*)d_out});
}
